// PointNetSetAbstractionMsg_20633022889985
// MI455X (gfx1250) — matrix operand order repaired and checked by reading, not compiled, not run on hardware
//
#include <hip/hip_runtime.h>

typedef _Float16 v8h  __attribute__((ext_vector_type(8)));
typedef _Float16 v16h __attribute__((ext_vector_type(16)));
typedef float    v8f  __attribute__((ext_vector_type(8)));

#define NPTS   16384
#define NSAMP  1024
#define NBATCH 8
#define C1     64

// ---------------------------------------------------------------------------
// Farthest point sampling: one 1024-thread block per batch, 16 points/thread
// kept in registers; per-iteration block argmax (wave shfl + LDS combine).
// ---------------------------------------------------------------------------
__global__ __launch_bounds__(1024) void fps_kernel(const float* __restrict__ xyz,
                                                   int* __restrict__ fps_idx,
                                                   float* __restrict__ new_xyz) {
  const int b   = blockIdx.x;
  const int tid = threadIdx.x;
  const float* base = xyz + (size_t)b * NPTS * 3;

  float px[16], py[16], pz[16], dist[16];
#pragma unroll
  for (int i = 0; i < 16; ++i) {
    int g = i * 1024 + tid;
    px[i] = base[g * 3 + 0];
    py[i] = base[g * 3 + 1];
    pz[i] = base[g * 3 + 2];
    dist[i] = 1e10f;
  }

  __shared__ float sc[3];
  __shared__ float wv[32];
  __shared__ int   wi[32];
  __shared__ int   scur;
  if (tid == 0) scur = 0;
  __syncthreads();

  for (int it = 0; it < NSAMP; ++it) {
    if (tid == 0) {
      int cur = scur;
      fps_idx[b * NSAMP + it] = cur;
      float x = base[cur * 3 + 0], y = base[cur * 3 + 1], z = base[cur * 3 + 2];
      sc[0] = x; sc[1] = y; sc[2] = z;
      float* nz = new_xyz + ((size_t)b * NSAMP + it) * 3;
      nz[0] = x; nz[1] = y; nz[2] = z;
    }
    __syncthreads();
    float cx = sc[0], cy = sc[1], cz = sc[2];

    float bv = -1.0f; int bi = 0;
#pragma unroll
    for (int i = 0; i < 16; ++i) {
      float dx = px[i] - cx, dy = py[i] - cy, dz = pz[i] - cz;
      float d = dx * dx + dy * dy + dz * dz;
      dist[i] = fminf(dist[i], d);
      int g = i * 1024 + tid;
      if (dist[i] > bv || (dist[i] == bv && g < bi)) { bv = dist[i]; bi = g; }
    }
    // wave32 argmax (tie-break: lowest index, matches jnp.argmax)
#pragma unroll
    for (int off = 16; off; off >>= 1) {
      float ov = __shfl_xor(bv, off);
      int   oi = __shfl_xor(bi, off);
      if (ov > bv || (ov == bv && oi < bi)) { bv = ov; bi = oi; }
    }
    int w = tid >> 5;
    if ((tid & 31) == 0) { wv[w] = bv; wi[w] = bi; }
    __syncthreads();
    if (w == 0) {
      bv = wv[tid & 31]; bi = wi[tid & 31];
#pragma unroll
      for (int off = 16; off; off >>= 1) {
        float ov = __shfl_xor(bv, off);
        int   oi = __shfl_xor(bi, off);
        if (ov > bv || (ov == bv && oi < bi)) { bv = ov; bi = oi; }
      }
      if (tid == 0) scur = bi;
    }
    __syncthreads();
  }
}

// ---------------------------------------------------------------------------
// Ball query: one wave per centroid; ordered compaction of the first K
// ascending in-radius indices via ballot + popcount, padded with the first.
// ---------------------------------------------------------------------------
__global__ void ballquery_kernel(const float* __restrict__ xyz,
                                 const float* __restrict__ new_xyz,
                                 int* __restrict__ idxout, int K, float r2) {
  int w    = blockIdx.x * (blockDim.x >> 5) + (threadIdx.x >> 5);
  int lane = threadIdx.x & 31;
  if (w >= NBATCH * NSAMP) return;
  int b = w >> 10;
  const float* q = new_xyz + (size_t)w * 3;
  float qx = q[0], qy = q[1], qz = q[2];
  const float* base = xyz + (size_t)b * NPTS * 3;
  int* out = idxout + (size_t)w * K;

  int cnt = 0, first = -1;
  for (int start = 0; start < NPTS; start += 32) {
    int i = start + lane;
    float dx = base[i * 3 + 0] - qx;
    float dy = base[i * 3 + 1] - qy;
    float dz = base[i * 3 + 2] - qz;
    bool in = (dx * dx + dy * dy + dz * dz) <= r2;
    unsigned m = (unsigned)__ballot(in);
    if (first < 0 && m) first = start + (__ffs((int)m) - 1);
    int pos = cnt + __popc(m & ((1u << lane) - 1u));
    if (in && pos < K) out[pos] = i;
    cnt += __popc(m);
    if (cnt >= K) break;
  }
  if (cnt < K) {
    for (int j = cnt + lane; j < K; j += 32) out[j] = first;
  }
}

// ---------------------------------------------------------------------------
// Gather grouped features into f16, Cin padded to 96 (cols 67..95 zero).
// Row m = ((b*1024)+s)*K + j ; cols: [0..2]=xyz[idx]-new_xyz, [3..66]=feature.
// ---------------------------------------------------------------------------
__global__ void gather_kernel(const float* __restrict__ xyz,
                              const float* __restrict__ feat,
                              const float* __restrict__ new_xyz,
                              const int* __restrict__ idx,
                              _Float16* __restrict__ X, int K, int total) {
  int gid = blockIdx.x * blockDim.x + threadIdx.x;
  if (gid >= total) return;                    // total = M * 12
  int m  = gid / 12;
  int c0 = (gid - m * 12) * 8;
  int bk = NSAMP * K;
  int b  = m / bk;
  int rem = m - b * bk;
  int s  = rem / K;
  int i  = idx[m];
  const float* p = xyz  + ((size_t)b * NPTS + i) * 3;
  const float* q = new_xyz + ((size_t)b * NSAMP + s) * 3;
  const float* f = feat + ((size_t)b * NPTS + i) * C1;
  v8h outv;
#pragma unroll
  for (int t = 0; t < 8; ++t) {
    int c = c0 + t;
    float v;
    if (c < 3)       v = p[c] - q[c];
    else if (c < 67) v = f[c - 3];
    else             v = 0.0f;
    outv[t] = (_Float16)v;
  }
  *(v8h*)(X + (size_t)m * 96 + c0) = outv;
}

// ---------------------------------------------------------------------------
// Zero-pad + convert one weight matrix to f16: Wp[n][k<cin]=W, else 0.
// ---------------------------------------------------------------------------
__global__ void wconv_kernel(const float* __restrict__ W, _Float16* __restrict__ Wp,
                             int cin, int cinpad, int total) {
  int gid = blockIdx.x * blockDim.x + threadIdx.x;
  if (gid >= total) return;
  int n = gid / cinpad;
  int k = gid - n * cinpad;
  Wp[gid] = (k < cin) ? (_Float16)W[(size_t)n * cin + k] : (_Float16)0.0f;
}

__global__ void zero_kernel(float* p, int n) {
  int gid = blockIdx.x * blockDim.x + threadIdx.x;
  if (gid < n) p[gid] = 0.0f;
}

// ---------------------------------------------------------------------------
// WMMA GEMM + BN-stats.  Y[M x COUT] = X[M x CINPAD] * W^T, f16 in, f32 acc.
// Register-blocked: one wave owns a 64x64 output block (4 M-tiles x 4 N-tiles,
// 128 f32 acc VGPRs).  Per K-step: 8 A loads + 8 B loads feed 16 independent
// WMMAs (1 load per WMMA, 4x A/B register reuse, latency hidden under XDL).
// A (16x32 f16) per-lane: row = m0+(L&15); halves 0-7 <- K kb..kb+7,
//                          halves 8-15 <- K kb+16..kb+23, kb = 8*(L>=16).
// B (32x16 f16) per-lane: col n = L&15; halves 0-7 <- K kb..kb+7,
//                          halves 8-15 <- K kb+16..kb+23.  Both are two global b128 loads.
// Per-channel sum / sumsq reduced shfl_xor(16) -> LDS ds_add -> global atomic.
// ---------------------------------------------------------------------------
template <int CINPAD, int COUT>
__global__ __launch_bounds__(256) void gemm_bn_kernel(
    const _Float16* __restrict__ X, const _Float16* __restrict__ W,
    _Float16* __restrict__ Y, float* __restrict__ sums, float* __restrict__ sqs,
    int M) {
  constexpr int KT   = CINPAD / 32;   // K-steps of 32
  constexpr int NGRP = COUT / 64;     // N wave-groups of 64 columns

  __shared__ float lsum[COUT];
  __shared__ float lsq[COUT];
  for (int c = threadIdx.x; c < COUT; c += blockDim.x) { lsum[c] = 0.0f; lsq[c] = 0.0f; }
  __syncthreads();

  int wave = blockIdx.x * (blockDim.x >> 5) + (threadIdx.x >> 5);
  int lane = threadIdx.x & 31;
  int mid  = wave / NGRP;
  int nid  = wave - mid * NGRP;
  int m0   = mid * 64;
  int n0   = nid * 64;

  if (m0 < M) {
    v8f acc[4][4] = {};
    const int hiA = (lane & 16) ? 8 : 0;    // A k-sub-base
    const int hiB = (lane & 16) ? 8 : 0;    // B k-sub-base
    const _Float16* xbase = X + (size_t)(m0 + (lane & 15)) * CINPAD + hiA;
    const _Float16* wbase = W + (size_t)(n0 + (lane & 15)) * CINPAD + hiB;
    __builtin_prefetch(xbase + (size_t)64 * CINPAD, 0, 1);

#pragma unroll
    for (int kt = 0; kt < KT; ++kt) {
      v16h a[4], bm[4];
#pragma unroll
      for (int mt = 0; mt < 4; ++mt) {
        const _Float16* p = xbase + (size_t)mt * 16 * CINPAD + kt * 32;
        *((v8h*)&a[mt])     = *(const v8h*)(p);
        *((v8h*)&a[mt] + 1) = *(const v8h*)(p + 16);
      }
#pragma unroll
      for (int nt = 0; nt < 4; ++nt) {
        const _Float16* p = wbase + (size_t)nt * 16 * CINPAD + kt * 32;
        *((v8h*)&bm[nt])     = *(const v8h*)(p);
        *((v8h*)&bm[nt] + 1) = *(const v8h*)(p + 16);
      }
#pragma unroll
      for (int mt = 0; mt < 4; ++mt) {
#pragma unroll
        for (int nt = 0; nt < 4; ++nt) {
          acc[mt][nt] = __builtin_amdgcn_wmma_f32_16x16x32_f16(
              false, a[mt], false, bm[nt], (short)0, acc[mt][nt], false, false);
        }
      }
    }

    // store f16 result + per-channel partial stats
    const int mrow = (lane & 16) ? 8 : 0;
#pragma unroll
    for (int nt = 0; nt < 4; ++nt) {
      float s = 0.0f, qq = 0.0f;
#pragma unroll
      for (int mt = 0; mt < 4; ++mt) {
#pragma unroll
        for (int r = 0; r < 8; ++r) {
          float v = acc[mt][nt][r];
          int mm = m0 + mt * 16 + r + mrow;
          Y[(size_t)mm * COUT + n0 + nt * 16 + (lane & 15)] = (_Float16)v;
          s += v; qq += v * v;
        }
      }
      s  += __shfl_xor(s, 16);
      qq += __shfl_xor(qq, 16);
      if (lane < 16) {
        atomicAdd(&lsum[n0 + nt * 16 + lane], s);
        atomicAdd(&lsq[n0 + nt * 16 + lane], qq);
      }
    }
  }
  __syncthreads();
  for (int c = threadIdx.x; c < COUT; c += blockDim.x) {
    atomicAdd(&sums[c], lsum[c]);
    atomicAdd(&sqs[c],  lsq[c]);
  }
}

// scale[c] = g*rsqrt(var+eps); bias[c] = beta - mean*scale.
// (Linear bias cancels exactly under BN mean-subtraction, so it is skipped.)
__global__ void bn_finalize_kernel(const float* __restrict__ sums,
                                   const float* __restrict__ sqs,
                                   const float* __restrict__ g,
                                   const float* __restrict__ be,
                                   float* __restrict__ scale,
                                   float* __restrict__ bias, int Cout, float invM) {
  int c = blockIdx.x * blockDim.x + threadIdx.x;
  if (c >= Cout) return;
  float mean = sums[c] * invM;
  float var  = sqs[c] * invM - mean * mean;
  float sc   = g[c] * rsqrtf(var + 1e-5f);
  scale[c] = sc;
  bias[c]  = be[c] - mean * sc;
}

// normalize + ReLU -> next layer's f16 activations
__global__ void act_kernel(const _Float16* __restrict__ Y, _Float16* __restrict__ X,
                           const float* __restrict__ scale,
                           const float* __restrict__ bias, int Cout, long long total) {
  long long gid = (long long)blockIdx.x * blockDim.x + threadIdx.x;
  if (gid >= total) return;
  int c = (int)(gid % Cout);
  float v = scale[c] * (float)Y[gid] + bias[c];
  X[gid] = (_Float16)fmaxf(v, 0.0f);
}

// normalize + ReLU + max over K neighbors -> f32 output slice
__global__ void maxpool_kernel(const _Float16* __restrict__ Y,
                               const float* __restrict__ scale,
                               const float* __restrict__ bias,
                               float* __restrict__ out, int K, int Cout,
                               int chanBase, int total) {
  int gid = blockIdx.x * blockDim.x + threadIdx.x;
  if (gid >= total) return;                 // total = B*N2*Cout
  int c  = gid % Cout;
  int bs = gid / Cout;
  const _Float16* yp = Y + (size_t)bs * K * Cout + c;
  float sc = scale[c], bi = bias[c];
  float m = 0.0f;                           // ReLU >= 0, so 0 is a valid floor
  for (int j = 0; j < K; ++j) {
    float v = sc * (float)yp[(size_t)j * Cout] + bi;
    m = fmaxf(m, v);
  }
  out[(size_t)bs * 384 + chanBase + c] = m;
}

// ---------------------------------------------------------------------------
// Host-side layer driver
// ---------------------------------------------------------------------------
template <int CINPAD, int COUT>
static inline void run_layer(const float* Wf, const float* g, const float* be,
                             const _Float16* Xin, _Float16* Yb, _Float16* Xnext,
                             _Float16* Wp, float* sums, float* sqs, float* scale,
                             float* biasv, int cinActual, int M, bool last,
                             float* poolOut, int K, int chanBase,
                             hipStream_t stream) {
  int wtotal = COUT * CINPAD;
  wconv_kernel<<<(wtotal + 255) / 256, 256, 0, stream>>>(Wf, Wp, cinActual, CINPAD, wtotal);
  zero_kernel<<<1, 256, 0, stream>>>(sums, 256);
  zero_kernel<<<1, 256, 0, stream>>>(sqs, 256);
  int waves = (M / 64) * (COUT / 64);       // one 64x64 block per wave
  gemm_bn_kernel<CINPAD, COUT><<<waves / 8, 256, 0, stream>>>(Xin, Wp, Yb, sums, sqs, M);
  bn_finalize_kernel<<<1, 256, 0, stream>>>(sums, sqs, g, be, scale, biasv, COUT,
                                            1.0f / (float)M);
  if (!last) {
    long long total = (long long)M * COUT;
    act_kernel<<<(unsigned)((total + 255) / 256), 256, 0, stream>>>(Yb, Xnext, scale,
                                                                    biasv, COUT, total);
  } else {
    int total = NBATCH * NSAMP * COUT;
    maxpool_kernel<<<(total + 255) / 256, 256, 0, stream>>>(Yb, scale, biasv, poolOut,
                                                            K, COUT, chanBase, total);
  }
}

extern "C" void kernel_launch(void* const* d_in, const int* in_sizes, int n_in,
                              void* d_out, int out_size, void* d_ws, size_t ws_size,
                              hipStream_t stream) {
  const float* xyz  = (const float*)d_in[0];
  const float* feat = (const float*)d_in[1];
  // params flatten: per branch, per layer: (W, b, gamma, beta)
  auto P = [&](int branch, int layer, int which) -> const float* {
    return (const float*)d_in[2 + branch * 12 + layer * 4 + which];
  };

  char* ws = (char*)d_ws;
  size_t off = 0;
  auto carve = [&](size_t bytes) -> char* {
    char* p = ws + off;
    off += (bytes + 255) & ~(size_t)255;
    return p;
  };
  int*      fps_idx = (int*)carve((size_t)NBATCH * NSAMP * 4);
  float*    nxyz    = (float*)carve((size_t)NBATCH * NSAMP * 3 * 4);
  int*      idx     = (int*)carve((size_t)NBATCH * NSAMP * 32 * 4);
  _Float16* Wp      = (_Float16*)carve((size_t)256 * 128 * 2);
  float*    sums    = (float*)carve(256 * 4);
  float*    sqs     = (float*)carve(256 * 4);
  float*    scale   = (float*)carve(256 * 4);
  float*    biasv   = (float*)carve(256 * 4);
  _Float16* Xb      = (_Float16*)carve((size_t)262144 * 128 * 2);
  _Float16* Yb      = (_Float16*)carve((size_t)262144 * 256 * 2);
  float*    out     = (float*)d_out;
  (void)in_sizes; (void)n_in; (void)out_size; (void)ws_size;

  // 1) farthest point sampling + new_xyz
  fps_kernel<<<NBATCH, 1024, 0, stream>>>(xyz, fps_idx, nxyz);

  // ---- branch 0: K=16, r=0.2, channels 96->64->64->128, out ch [0,128) ----
  {
    const int K = 16;
    const float r2 = 0.2f * 0.2f;
    const int M = NBATCH * NSAMP * K;
    ballquery_kernel<<<(NBATCH * NSAMP) / 8, 256, 0, stream>>>(xyz, nxyz, idx, K, r2);
    gather_kernel<<<(M * 12 + 255) / 256, 256, 0, stream>>>(xyz, feat, nxyz, idx, Xb, K,
                                                            M * 12);
    run_layer<96, 64>(P(0, 0, 0), P(0, 0, 2), P(0, 0, 3), Xb, Yb, Xb, Wp, sums, sqs,
                      scale, biasv, 67, M, false, nullptr, K, 0, stream);
    run_layer<64, 64>(P(0, 1, 0), P(0, 1, 2), P(0, 1, 3), Xb, Yb, Xb, Wp, sums, sqs,
                      scale, biasv, 64, M, false, nullptr, K, 0, stream);
    run_layer<64, 128>(P(0, 2, 0), P(0, 2, 2), P(0, 2, 3), Xb, Yb, nullptr, Wp, sums,
                       sqs, scale, biasv, 64, M, true, out, K, 0, stream);
  }

  // ---- branch 1: K=32, r=0.4, channels 96->128->128->256, out ch [128,384) ----
  {
    const int K = 32;
    const float r2 = 0.4f * 0.4f;
    const int M = NBATCH * NSAMP * K;
    ballquery_kernel<<<(NBATCH * NSAMP) / 8, 256, 0, stream>>>(xyz, nxyz, idx, K, r2);
    gather_kernel<<<(M * 12 + 255) / 256, 256, 0, stream>>>(xyz, feat, nxyz, idx, Xb, K,
                                                            M * 12);
    run_layer<96, 128>(P(1, 0, 0), P(1, 0, 2), P(1, 0, 3), Xb, Yb, Xb, Wp, sums, sqs,
                       scale, biasv, 67, M, false, nullptr, K, 128, stream);
    run_layer<128, 128>(P(1, 1, 0), P(1, 1, 2), P(1, 1, 3), Xb, Yb, Xb, Wp, sums, sqs,
                        scale, biasv, 128, M, false, nullptr, K, 128, stream);
    run_layer<128, 256>(P(1, 2, 0), P(1, 2, 2), P(1, 2, 3), Xb, Yb, nullptr, Wp, sums,
                        sqs, scale, biasv, 128, M, true, out, K, 128, stream);
  }
}
